// VQEmbedding_59691455480165
// MI455X (gfx1250) — matrix operand order repaired and checked by reading, not compiled, not run on hardware
//
#include <hip/hip_runtime.h>

typedef __attribute__((ext_vector_type(16))) __bf16 v16bf;
typedef __attribute__((ext_vector_type(8)))  float  v8f;

#define N_ROWS  131072
#define DIM     256
#define KCODES  1024
#define NSTEPS  (KCODES / 32)     // 32 staging steps, 2 column tiles each
#define DECAY_F 0.99f
#define EPS_F   1e-5f
#define CCOST_F 1.0f

#if defined(__AMDGCN__) && __has_builtin(__builtin_amdgcn_global_load_async_to_lds_b128) && __has_builtin(__builtin_amdgcn_s_wait_asynccnt)
#define VQ_ASYNC 1
typedef int vq_v4i __attribute__((vector_size(16)));
typedef __attribute__((address_space(1))) vq_v4i gv4i_t;
typedef __attribute__((address_space(3))) vq_v4i lv4i_t;
#else
#define VQ_ASYNC 0
#endif

union Frag {
    unsigned short s[16];
    uint4          u[2];
    v16bf          v;
};

__device__ __forceinline__ unsigned short f2bf(float f) {
    unsigned int u = __builtin_bit_cast(unsigned int, f);
    u += 0x7FFFu + ((u >> 16) & 1u);            // round-to-nearest-even
    return (unsigned short)(u >> 16);
}

// ---------------------------------------------------------------------------
// Kernel 0: embedding f32 -> bf16 workspace copy + per-code squared L2 norm
// ---------------------------------------------------------------------------
__global__ void vq_prep(const float* __restrict__ emb,
                        unsigned short* __restrict__ emb_bf,
                        float* __restrict__ e2) {
    int k = blockIdx.x;
    int d = threadIdx.x;
    float v = emb[(size_t)k * DIM + d];
    emb_bf[(size_t)k * DIM + d] = f2bf(v);
    __shared__ float red[256];
    red[d] = v * v;
    __syncthreads();
    #pragma unroll
    for (int s = 128; s > 0; s >>= 1) {
        if (d < s) red[d] += red[d + s];
        __syncthreads();
    }
    if (d == 0) e2[k] = red[0];
}

// ---------------------------------------------------------------------------
// Kernel 1: zero accumulators
// ---------------------------------------------------------------------------
__global__ void vq_zero(float* __restrict__ p, int n) {
    int i = blockIdx.x * 256 + threadIdx.x;
    if (i < n) p[i] = 0.0f;
}

// ---------------------------------------------------------------------------
// Kernel 2: distances + argmin via v_wmma_f32_16x16x32_bf16
// 256 threads = 8 waves; block owns 128 rows, wave owns 16 rows.
// Double-buffered 32-code LDS tiles; async global->LDS staging overlaps
// the 16 WMMAs (2 independent accumulator chains) of the current tile.
// ---------------------------------------------------------------------------
__global__ void __launch_bounds__(256)
vq_argmin(const float* __restrict__ x,
          const unsigned short* __restrict__ emb_bf,
          const float* __restrict__ e2,
          int* __restrict__ idx_ws,
          float* __restrict__ out_idx) {
    __shared__ __align__(16) unsigned short lds_b[2][32 * DIM];  // 2 x 16 KB

    const int tid   = threadIdx.x;
    const int wave  = tid >> 5;
    const int lane  = tid & 31;
    const int nlane = lane & 15;               // column within 16-wide tile
    const int hi    = (lane >> 4) & 1;         // lane-half selector
    const int rowbase = blockIdx.x * 128 + wave * 16;

    // ---- Load & convert this wave's 16x256 A panel (bf16 fragments) ----
    Frag af[8];
    {
        const float* xrow = x + (size_t)(rowbase + nlane) * DIM;
        const int sel8 = hi * 8;
        #pragma unroll
        for (int kb = 0; kb < 8; ++kb) {
            const float* p0 = xrow + kb * 32 + sel8;        // K = kb*32 + sel8 .. +7
            const float* p1 = xrow + kb * 32 + 16 + sel8;   // K = kb*32+16+sel8 .. +7
            float4 q0 = *(const float4*)(p0);
            float4 q1 = *(const float4*)(p0 + 4);
            float4 q2 = *(const float4*)(p1);
            float4 q3 = *(const float4*)(p1 + 4);
            af[kb].s[0]  = f2bf(q0.x); af[kb].s[1]  = f2bf(q0.y);
            af[kb].s[2]  = f2bf(q0.z); af[kb].s[3]  = f2bf(q0.w);
            af[kb].s[4]  = f2bf(q1.x); af[kb].s[5]  = f2bf(q1.y);
            af[kb].s[6]  = f2bf(q1.z); af[kb].s[7]  = f2bf(q1.w);
            af[kb].s[8]  = f2bf(q2.x); af[kb].s[9]  = f2bf(q2.y);
            af[kb].s[10] = f2bf(q2.z); af[kb].s[11] = f2bf(q2.w);
            af[kb].s[12] = f2bf(q3.x); af[kb].s[13] = f2bf(q3.y);
            af[kb].s[14] = f2bf(q3.z); af[kb].s[15] = f2bf(q3.w);
        }
    }

    float minv[8];
    int   mini[8];
    #pragma unroll
    for (int i = 0; i < 8; ++i) { minv[i] = 3.0e38f; mini[i] = 0; }

    // each thread stages 32 halves (64 B) of the 32x256-half tile
    const int stage_h = tid * 32;

    // ---- stage tile 0 into buffer 0 ----
    {
        const unsigned short* src = emb_bf + stage_h;       // step 0 base
        unsigned short* dst = &lds_b[0][stage_h];
#if VQ_ASYNC
        gv4i_t* gs = (gv4i_t*)(uintptr_t)src;
        lv4i_t* ld = (lv4i_t*)(unsigned int)(uintptr_t)dst;
        __builtin_amdgcn_global_load_async_to_lds_b128(gs, ld, 0,  0);
        __builtin_amdgcn_global_load_async_to_lds_b128(gs, ld, 16, 0);
        __builtin_amdgcn_global_load_async_to_lds_b128(gs, ld, 32, 0);
        __builtin_amdgcn_global_load_async_to_lds_b128(gs, ld, 48, 0);
        __builtin_amdgcn_s_wait_asynccnt(0);
#else
        const uint4* s4 = (const uint4*)src;
        uint4 a0 = s4[0], a1 = s4[1], a2 = s4[2], a3 = s4[3];
        uint4* d4 = (uint4*)dst;
        d4[0] = a0; d4[1] = a1; d4[2] = a2; d4[3] = a3;
#endif
    }
    __syncthreads();

    for (int nt = 0; nt < NSTEPS; ++nt) {
        const int cur = nt & 1;

        // ---- kick off async stage of the NEXT 32-code tile ----
        if (nt + 1 < NSTEPS) {
            const unsigned short* src =
                emb_bf + ((size_t)(nt + 1) << 13) + stage_h;   // (nt+1)*32*256
            unsigned short* dst = &lds_b[cur ^ 1][stage_h];
#if VQ_ASYNC
            gv4i_t* gs = (gv4i_t*)(uintptr_t)src;
            lv4i_t* ld = (lv4i_t*)(unsigned int)(uintptr_t)dst;
            __builtin_amdgcn_global_load_async_to_lds_b128(gs, ld, 0,  0);
            __builtin_amdgcn_global_load_async_to_lds_b128(gs, ld, 16, 0);
            __builtin_amdgcn_global_load_async_to_lds_b128(gs, ld, 32, 0);
            __builtin_amdgcn_global_load_async_to_lds_b128(gs, ld, 48, 0);
#else
            const uint4* s4 = (const uint4*)src;
            uint4 a0 = s4[0], a1 = s4[1], a2 = s4[2], a3 = s4[3];
            uint4* d4 = (uint4*)dst;
            d4[0] = a0; d4[1] = a1; d4[2] = a2; d4[3] = a3;
#endif
        }

        // ---- two 16x16 dot tiles: 16 WMMAs, 2 independent chains ----
        v8f c0 = {0.f, 0.f, 0.f, 0.f, 0.f, 0.f, 0.f, 0.f};
        v8f c1 = {0.f, 0.f, 0.f, 0.f, 0.f, 0.f, 0.f, 0.f};
        const unsigned short* b0p = &lds_b[cur][(nlane << 8) + (hi << 3)];
        const unsigned short* b1p = b0p + (16 << 8);   // codes +16
        #pragma unroll
        for (int kb = 0; kb < 8; ++kb) {
            Frag b0, b1;
            const uint4* p0 = (const uint4*)(b0p + kb * 32);
            const uint4* p1 = (const uint4*)(b1p + kb * 32);
            b0.u[0] = p0[0]; b0.u[1] = p0[2];
            b1.u[0] = p1[0]; b1.u[1] = p1[2];
            c0 = __builtin_amdgcn_wmma_f32_16x16x32_bf16(
                     false, af[kb].v, false, b0.v, (short)0, c0, false, false);
            c1 = __builtin_amdgcn_wmma_f32_16x16x32_bf16(
                     false, af[kb].v, false, b1.v, (short)0, c1, false, false);
        }

        // ---- score = e2 - 2*dot (x^2 is row-constant; irrelevant to argmin)
        const int n0 = nt * 32 + nlane;
        const int n1 = n0 + 16;
        const float e0 = e2[n0];
        const float e1 = e2[n1];
        #pragma unroll
        for (int i = 0; i < 8; ++i) {
            float s0 = fmaf(-2.0f, c0[i], e0);
            if (s0 < minv[i]) { minv[i] = s0; mini[i] = n0; }
            float s1 = fmaf(-2.0f, c1[i], e1);
            if (s1 < minv[i]) { minv[i] = s1; mini[i] = n1; }
        }

#if VQ_ASYNC
        __builtin_amdgcn_s_wait_asynccnt(0);
#endif
        __syncthreads();   // next buffer ready for all; current free to overwrite
    }

    // ---- cross-lane argmin within each 16-lane group (matches C layout) ----
    #pragma unroll
    for (int i = 0; i < 8; ++i) {
        float v  = minv[i];
        int   id = mini[i];
        #pragma unroll
        for (int off = 1; off < 16; off <<= 1) {
            float ov = __shfl_xor(v, off, 32);
            int   oi = __shfl_xor(id, off, 32);
            if (ov < v || (ov == v && oi < id)) { v = ov; id = oi; }
        }
        if (nlane == 0) {
            int m = rowbase + (hi << 3) + i;   // lanes 0..15 -> rows i; 16..31 -> i+8
            idx_ws[m]  = id;
            out_idx[m] = (float)id;
        }
    }
}

// ---------------------------------------------------------------------------
// Kernel 3: gather quantized, straight-through output, loss + segment sums
// ---------------------------------------------------------------------------
__global__ void vq_scatter(const float* __restrict__ x,
                           const float* __restrict__ emb,
                           const int* __restrict__ idx,
                           float* __restrict__ out_q,
                           float* __restrict__ enc_sum,
                           float* __restrict__ emb_sum,
                           float* __restrict__ loss_acc) {
    const int d = threadIdx.x;
    const int row0 = blockIdx.x * 32;
    float lsum = 0.0f;
    for (int r = 0; r < 32; ++r) {
        const int row = row0 + r;
        const int k = idx[row];
        const float xv = x[(size_t)row * DIM + d];
        const float q  = emb[(size_t)k * DIM + d];
        out_q[(size_t)row * DIM + d] = xv + (q - xv);   // straight-through value
        const float diff = xv - q;
        lsum += diff * diff;
        atomicAdd(&emb_sum[(size_t)k * DIM + d], xv);
        if (d == 0) atomicAdd(&enc_sum[k], 1.0f);
    }
    __shared__ float red[256];
    red[d] = lsum;
    __syncthreads();
    #pragma unroll
    for (int s = 128; s > 0; s >>= 1) {
        if (d < s) red[d] += red[d + s];
        __syncthreads();
    }
    if (d == 0) atomicAdd(loss_acc, red[0]);
}

// ---------------------------------------------------------------------------
// Kernel 4: new_cluster_size + global n reduction
// ---------------------------------------------------------------------------
__global__ void vq_ema1(const float* __restrict__ cluster_size,
                        const float* __restrict__ enc_sum,
                        float* __restrict__ out_ncs,
                        float* __restrict__ n_acc) {
    const int k = blockIdx.x * 256 + threadIdx.x;
    float ncs = cluster_size[k] * DECAY_F + enc_sum[k] * (1.0f - DECAY_F);
    out_ncs[k] = ncs;
    __shared__ float red[256];
    red[threadIdx.x] = ncs;
    __syncthreads();
    #pragma unroll
    for (int s = 128; s > 0; s >>= 1) {
        if (threadIdx.x < s) red[threadIdx.x] += red[threadIdx.x + s];
        __syncthreads();
    }
    if (threadIdx.x == 0) atomicAdd(n_acc, red[0]);
}

// ---------------------------------------------------------------------------
// Kernel 5: new_embedding_avg, normalized new_embedding, final loss scalar
// ---------------------------------------------------------------------------
__global__ void vq_ema2(const float* __restrict__ emb_avg,
                        const float* __restrict__ emb_sum,
                        const float* __restrict__ out_ncs,
                        const float* __restrict__ scalars,   // [0]=loss sum, [1]=n
                        float* __restrict__ out_ne,
                        float* __restrict__ out_nea,
                        float* __restrict__ out_loss) {
    const int i = blockIdx.x * 256 + threadIdx.x;   // over K*D
    const int k = i >> 8;
    float nea = emb_avg[i] * DECAY_F + emb_sum[i] * (1.0f - DECAY_F);
    out_nea[i] = nea;
    const float n   = scalars[1];
    const float ncs = out_ncs[k];
    const float cs  = (ncs + EPS_F) / (n + (float)KCODES * EPS_F) * n;
    out_ne[i] = nea / cs;
    if (i == 0)
        out_loss[0] = scalars[0] * (CCOST_F / ((float)N_ROWS * (float)DIM));
}

// ---------------------------------------------------------------------------
extern "C" void kernel_launch(void* const* d_in, const int* in_sizes, int n_in,
                              void* d_out, int out_size, void* d_ws, size_t ws_size,
                              hipStream_t stream) {
    const float* x     = (const float*)d_in[0];   // [N, D]
    const float* emb   = (const float*)d_in[1];   // [K, D]
    const float* csize = (const float*)d_in[2];   // [K]
    const float* eavg  = (const float*)d_in[3];   // [K, D]

    float* out = (float*)d_out;
    // output layout (flat, in reference return order)
    float* out_q    = out;                                        // N*D
    float* out_loss = out + (size_t)N_ROWS * DIM;                 // 1
    float* out_idx  = out_loss + 1;                               // N
    float* out_ne   = out_idx + N_ROWS;                           // K*D
    float* out_ncs  = out_ne + (size_t)KCODES * DIM;              // K
    float* out_nea  = out_ncs + KCODES;                           // K*D

    // workspace layout
    float* ws = (float*)d_ws;
    float*          ws_e2    = ws;                                              // K
    unsigned short* ws_embbf = (unsigned short*)(ws + KCODES);                  // K*D bf16
    int*            ws_idx   = (int*)(ws + KCODES + (KCODES * DIM) / 2);        // N
    float*          ws_enc   = ws + KCODES + (KCODES * DIM) / 2 + N_ROWS;       // K
    float*          ws_esum  = ws_enc + KCODES;                                 // K*D
    float*          ws_scal  = ws_esum + (size_t)KCODES * DIM;                  // [loss, n]

    const int zero_n = KCODES + KCODES * DIM + 2;  // enc_sum | emb_sum | scalars

    vq_prep<<<KCODES, 256, 0, stream>>>(emb, ws_embbf, ws_e2);
    vq_zero<<<(zero_n + 255) / 256, 256, 0, stream>>>(ws_enc, zero_n);
    vq_argmin<<<N_ROWS / 128, 256, 0, stream>>>(x, ws_embbf, ws_e2, ws_idx, out_idx);
    vq_scatter<<<N_ROWS / 32, 256, 0, stream>>>(x, emb, ws_idx, out_q,
                                                ws_enc, ws_esum, ws_scal);
    vq_ema1<<<KCODES / 256, 256, 0, stream>>>(csize, ws_enc, out_ncs, ws_scal + 1);
    vq_ema2<<<(KCODES * DIM) / 256, 256, 0, stream>>>(eavg, ws_esum, out_ncs,
                                                      ws_scal, out_ne, out_nea, out_loss);
}
